// acLSTM_35811437314836
// MI455X (gfx1250) — matrix operand order repaired and checked by reading, not compiled, not run on hardware
//
#include <hip/hip_runtime.h>
#include <cstdint>
#include <cstddef>

// ---------------- problem constants (match reference) ----------------
#define BATCH 128
#define TINIT 50
#define NGEN  100
#define INF   171            // raw input features
#define KXP   192            // x padded to multiple of 32
#define HD    1024           // hidden size
#define OUTF  171            // decoder out features
#define NDEC  192            // decoder N padded
#define SROW  (KXP + 3*HD)   // 3264: [x | h0 | h1 | h2] row stride (bf16 elems)
#define G4H   (4*HD)         // 4096 gate rows

typedef __bf16 v16bf __attribute__((ext_vector_type(16)));
typedef __bf16 v8bf  __attribute__((ext_vector_type(8)));
typedef float  v8f   __attribute__((ext_vector_type(8)));

__device__ __forceinline__ unsigned short f2bf(float f) {
  unsigned int u = __float_as_uint(f);
  unsigned int r = u + 0x7FFFu + ((u >> 16) & 1u);   // round-to-nearest-even
  return (unsigned short)(r >> 16);
}
__device__ __forceinline__ float sigm(float x) { return 1.0f / (1.0f + __expf(-x)); }

// ---------------- utility kernels ----------------
__global__ void zero_u32(uint32_t* __restrict__ p, int n) {
  for (int i = blockIdx.x * blockDim.x + threadIdx.x; i < n; i += gridDim.x * blockDim.x)
    p[i] = 0u;
}

// Pack [w_ih | w_hh] fp32 -> bf16 [4H x Kc] row-major, zero-padding cols KxReal..Kx.
// bias = b_ih + b_hh.
__global__ void pack_cell(const float* __restrict__ wih, const float* __restrict__ whh,
                          const float* __restrict__ bih, const float* __restrict__ bhh,
                          unsigned short* __restrict__ Wp, float* __restrict__ bias,
                          int KxReal, int Kx, int Kc) {
  int total = G4H * Kc;
  for (int i = blockIdx.x * blockDim.x + threadIdx.x; i < total; i += gridDim.x * blockDim.x) {
    int j = i / Kc, k = i - j * Kc;
    float v;
    if (k < Kx) v = (k < KxReal) ? wih[j * KxReal + k] : 0.0f;
    else        v = whh[j * HD + (k - Kx)];
    Wp[i] = f2bf(v);
  }
  for (int i = blockIdx.x * blockDim.x + threadIdx.x; i < G4H; i += gridDim.x * blockDim.x)
    bias[i] = bih[i] + bhh[i];
}

__global__ void pack_dec(const float* __restrict__ wdec, const float* __restrict__ bdec,
                         unsigned short* __restrict__ Wp, float* __restrict__ bp) {
  int total = NDEC * HD;
  for (int i = blockIdx.x * blockDim.x + threadIdx.x; i < total; i += gridDim.x * blockDim.x) {
    int n = i / HD, k = i - n * HD;
    Wp[i] = f2bf(n < OUTF ? wdec[n * HD + k] : 0.0f);
  }
  for (int i = blockIdx.x * blockDim.x + threadIdx.x; i < NDEC; i += gridDim.x * blockDim.x)
    bp[i] = (i < OUTF) ? bdec[i] : 0.0f;
}

__global__ void load_frame(const float* __restrict__ seq, int t, unsigned short* __restrict__ xdst) {
  int total = BATCH * KXP;
  for (int i = blockIdx.x * blockDim.x + threadIdx.x; i < total; i += gridDim.x * blockDim.x) {
    int b = i / KXP, k = i - b * KXP;
    float v = (k < INF) ? seq[(b * TINIT + t) * INF + k] : 0.0f;
    xdst[(size_t)b * SROW + k] = f2bf(v);
  }
}

// ---------------- WMMA fragment helpers ----------------
// A fragment 16x32 bf16 from row-major A: lane l -> row m=l&15; contiguous
// k0+(l>>4)*16 .. +15 (matches B fragment k order).
__device__ __forceinline__ v16bf load_a_frag(const unsigned short* rowp, int hi) {
  return *(const v16bf*)(const void*)(rowp + hi * 16);
}
// B fragment 32x16 bf16: one contiguous 32B load per lane from row-major [N x K] weights.
__device__ __forceinline__ v16bf load_b_frag(const unsigned short* rowp, int hi) {
  return *(const v16bf*)(const void*)(rowp + hi * 16);
}
#define WMMA_BF16(A, B, C) \
  __builtin_amdgcn_wmma_f32_16x16x32_bf16(false, (A), false, (B), (short)0, (C), false, false)

// ---------------- fused LSTM cell ----------------
// grid = 32 blocks x 256 threads (8 waves). Each wave owns a 32(M)x16(N) tile of ALL
// four gates: 8 accumulators. B fragments are reused across the two M-tiles
// (0.75 loads per WMMA), A fragments across the four gates.
__global__ __launch_bounds__(256) void lstm_cell_wmma(
    const unsigned short* __restrict__ Ax, const unsigned short* __restrict__ Ah,
    int Kx, int Kc,
    const unsigned short* __restrict__ Wp, const float* __restrict__ bias,
    float* __restrict__ cst, unsigned short* __restrict__ hOut) {
  const int lane = threadIdx.x & 31;
  const int wave = threadIdx.x >> 5;
  const int mp     = blockIdx.x & 3;                 // 4 M-pairs (B=128, 32 rows each)
  const int n_tile = (blockIdx.x >> 2) * 8 + wave;   // 64 N tiles (H=1024)
  const int ln = lane & 15;
  const int hi = lane >> 4;

  const size_t gstride = (size_t)HD * Kc;
  const unsigned short* ax0 = Ax + (size_t)(mp * 32 + ln) * SROW;
  const unsigned short* ax1 = ax0 + (size_t)16 * SROW;
  const unsigned short* ah0 = Ah + (size_t)(mp * 32 + ln) * SROW;
  const unsigned short* ah1 = ah0 + (size_t)16 * SROW;
  const unsigned short* wI = Wp + (size_t)(n_tile * 16 + ln) * Kc;
  const unsigned short* wF = wI + gstride;
  const unsigned short* wG = wF + gstride;
  const unsigned short* wO = wG + gstride;

  const v8f z = {0.f, 0.f, 0.f, 0.f, 0.f, 0.f, 0.f, 0.f};
  v8f aI0 = z, aI1 = z, aF0 = z, aF1 = z, aG0 = z, aG1 = z, aO0 = z, aO1 = z;

  // ---- x-part of K (uniform pointers, no per-iter select) ----
  for (int k0 = 0; k0 < Kx; k0 += 32) {
    const v16bf a0 = load_a_frag(ax0 + k0, hi);
    const v16bf a1 = load_a_frag(ax1 + k0, hi);
    v16bf b;
    b = load_b_frag(wI + k0, hi); aI0 = WMMA_BF16(a0, b, aI0); aI1 = WMMA_BF16(a1, b, aI1);
    b = load_b_frag(wF + k0, hi); aF0 = WMMA_BF16(a0, b, aF0); aF1 = WMMA_BF16(a1, b, aF1);
    b = load_b_frag(wG + k0, hi); aG0 = WMMA_BF16(a0, b, aG0); aG1 = WMMA_BF16(a1, b, aG1);
    b = load_b_frag(wO + k0, hi); aO0 = WMMA_BF16(a0, b, aO0); aO1 = WMMA_BF16(a1, b, aO1);
  }
  // ---- h-part of K ----
  const unsigned short* wIh = wI + Kx;
  const unsigned short* wFh = wF + Kx;
  const unsigned short* wGh = wG + Kx;
  const unsigned short* wOh = wO + Kx;
  const int Kh = Kc - Kx;                            // == HD
  for (int k0 = 0; k0 < Kh; k0 += 32) {
    const v16bf a0 = load_a_frag(ah0 + k0, hi);
    const v16bf a1 = load_a_frag(ah1 + k0, hi);
    v16bf b;
    b = load_b_frag(wIh + k0, hi); aI0 = WMMA_BF16(a0, b, aI0); aI1 = WMMA_BF16(a1, b, aI1);
    b = load_b_frag(wFh + k0, hi); aF0 = WMMA_BF16(a0, b, aF0); aF1 = WMMA_BF16(a1, b, aF1);
    b = load_b_frag(wGh + k0, hi); aG0 = WMMA_BF16(a0, b, aG0); aG1 = WMMA_BF16(a1, b, aG1);
    b = load_b_frag(wOh + k0, hi); aO0 = WMMA_BF16(a0, b, aO0); aO1 = WMMA_BF16(a1, b, aO1);
  }

  // ---- epilogue: LSTM activations (gate order i, f, g, o) ----
  const int n_loc = n_tile * 16 + ln;
  const float bI = bias[n_loc];
  const float bF = bias[n_loc + HD];
  const float bG = bias[n_loc + 2 * HD];
  const float bO = bias[n_loc + 3 * HD];
#pragma unroll
  for (int mt = 0; mt < 2; ++mt) {
    const v8f gI = mt ? aI1 : aI0;
    const v8f gF = mt ? aF1 : aF0;
    const v8f gG = mt ? aG1 : aG0;
    const v8f gO = mt ? aO1 : aO0;
#pragma unroll
    for (int v = 0; v < 8; ++v) {
      const int m = mp * 32 + mt * 16 + hi * 8 + v;  // C layout: lanes>=16 hold M=8..15
      const float gi = gI[v] + bI;
      const float gf = gF[v] + bF;
      const float gg = gG[v] + bG;
      const float go = gO[v] + bO;
      const float cOld = cst[m * HD + n_loc];
      const float cNew = sigm(gf) * cOld + sigm(gi) * tanhf(gg);
      const float hNew = sigm(go) * tanhf(cNew);
      cst[m * HD + n_loc] = cNew;
      hOut[(size_t)m * SROW + n_loc] = f2bf(hNew);
    }
  }
}

// ---------------- decoder: out = h2 @ w_dec^T + b, dual write ----------------
// 48 waves = 4 M-pairs x 12 N tiles; grid = 6 blocks x 256 threads.
__global__ __launch_bounds__(256) void decoder_wmma(
    const unsigned short* __restrict__ Ah, const unsigned short* __restrict__ Wp,
    const float* __restrict__ bp, float* __restrict__ outBase,
    unsigned short* __restrict__ xNext, int writeOut, int tGen) {
  const int wglob = blockIdx.x * 8 + (threadIdx.x >> 5);
  const int lane  = threadIdx.x & 31;
  const int mp     = wglob & 3;
  const int n_tile = wglob >> 2;                     // 0..11
  const int ln = lane & 15;
  const int hi = lane >> 4;

  const unsigned short* a0r = Ah + (size_t)(mp * 32 + ln) * SROW;
  const unsigned short* a1r = a0r + (size_t)16 * SROW;
  const unsigned short* wr  = Wp + (size_t)(n_tile * 16 + ln) * HD;

  const v8f z = {0.f, 0.f, 0.f, 0.f, 0.f, 0.f, 0.f, 0.f};
  v8f acc0 = z, acc1 = z;
  for (int k0 = 0; k0 < HD; k0 += 32) {
    const v16bf a0 = load_a_frag(a0r + k0, hi);
    const v16bf a1 = load_a_frag(a1r + k0, hi);
    const v16bf b  = load_b_frag(wr + k0, hi);
    acc0 = WMMA_BF16(a0, b, acc0);
    acc1 = WMMA_BF16(a1, b, acc1);
  }

  const int n = n_tile * 16 + ln;
  const float bb = bp[n];
#pragma unroll
  for (int mt = 0; mt < 2; ++mt) {
    const v8f acc = mt ? acc1 : acc0;
#pragma unroll
    for (int v = 0; v < 8; ++v) {
      const int m = mp * 32 + mt * 16 + hi * 8 + v;
      const float val = acc[v] + bb;
      xNext[(size_t)m * SROW + n] = f2bf(val);       // padded cols produce exact 0
      if (writeOut && n < OUTF)
        outBase[((size_t)m * NGEN + tGen) * OUTF + n] = val;
    }
  }
}

// ---------------- host orchestration ----------------
extern "C" void kernel_launch(void* const* d_in, const int* in_sizes, int n_in,
                              void* d_out, int out_size, void* d_ws, size_t ws_size,
                              hipStream_t stream) {
  (void)in_sizes; (void)n_in; (void)out_size; (void)ws_size;
  const float* seq   = (const float*)d_in[0];
  // d_in[1] = generate_frames_number (fixed = NGEN)
  const float* w_ih1 = (const float*)d_in[2];
  const float* w_hh1 = (const float*)d_in[3];
  const float* b_ih1 = (const float*)d_in[4];
  const float* b_hh1 = (const float*)d_in[5];
  const float* w_ih2 = (const float*)d_in[6];
  const float* w_hh2 = (const float*)d_in[7];
  const float* b_ih2 = (const float*)d_in[8];
  const float* b_hh2 = (const float*)d_in[9];
  const float* w_ih3 = (const float*)d_in[10];
  const float* w_hh3 = (const float*)d_in[11];
  const float* b_ih3 = (const float*)d_in[12];
  const float* b_hh3 = (const float*)d_in[13];
  const float* w_dec = (const float*)d_in[14];
  const float* b_dec = (const float*)d_in[15];
  float* out = (float*)d_out;

  uint8_t* ws = (uint8_t*)d_ws;
  auto alloc = [&](size_t bytes) -> uint8_t* {
    uint8_t* p = ws;
    ws += (bytes + 255) & ~(size_t)255;
    return p;
  };
  const int Kc1 = KXP + HD;     // 1216
  const int Kc23 = 2 * HD;      // 2048
  unsigned short* Wp1 = (unsigned short*)alloc((size_t)G4H * Kc1  * 2);
  unsigned short* Wp2 = (unsigned short*)alloc((size_t)G4H * Kc23 * 2);
  unsigned short* Wp3 = (unsigned short*)alloc((size_t)G4H * Kc23 * 2);
  unsigned short* Wpd = (unsigned short*)alloc((size_t)NDEC * HD * 2);
  float* bias1 = (float*)alloc(G4H * 4);
  float* bias2 = (float*)alloc(G4H * 4);
  float* bias3 = (float*)alloc(G4H * 4);
  float* biasd = (float*)alloc(NDEC * 4);
  uint8_t* actBase = ws;        // contiguous activation region for one-shot zeroing
  unsigned short* hbuf0 = (unsigned short*)alloc((size_t)BATCH * SROW * 2);
  unsigned short* hbuf1 = (unsigned short*)alloc((size_t)BATCH * SROW * 2);
  float* c1 = (float*)alloc((size_t)BATCH * HD * 4);
  float* c2 = (float*)alloc((size_t)BATCH * HD * 4);
  float* c3 = (float*)alloc((size_t)BATCH * HD * 4);
  const size_t actBytes = (size_t)(ws - actBase);

  zero_u32<<<2048, 256, 0, stream>>>((uint32_t*)actBase, (int)(actBytes / 4));

  pack_cell<<<4096, 256, 0, stream>>>(w_ih1, w_hh1, b_ih1, b_hh1, Wp1, bias1, INF, KXP, Kc1);
  pack_cell<<<4096, 256, 0, stream>>>(w_ih2, w_hh2, b_ih2, b_hh2, Wp2, bias2, HD, HD, Kc23);
  pack_cell<<<4096, 256, 0, stream>>>(w_ih3, w_hh3, b_ih3, b_hh3, Wp3, bias3, HD, HD, Kc23);
  pack_dec<<<768, 256, 0, stream>>>(w_dec, b_dec, Wpd, biasd);

  load_frame<<<96, 256, 0, stream>>>(seq, 0, hbuf0);

  unsigned short* bufs[2] = {hbuf0, hbuf1};
  for (int t = 0; t < TINIT + NGEN; ++t) {
    unsigned short* prev = bufs[t & 1];
    unsigned short* cur  = bufs[(t + 1) & 1];

    lstm_cell_wmma<<<32, 256, 0, stream>>>(
        prev + 0, prev + KXP, KXP, Kc1, Wp1, bias1, c1, cur + KXP);
    lstm_cell_wmma<<<32, 256, 0, stream>>>(
        cur + KXP, prev + KXP + HD, HD, Kc23, Wp2, bias2, c2, cur + KXP + HD);
    lstm_cell_wmma<<<32, 256, 0, stream>>>(
        cur + KXP + HD, prev + KXP + 2 * HD, HD, Kc23, Wp3, bias3, c3, cur + KXP + 2 * HD);

    if (t < TINIT - 1) {
      load_frame<<<96, 256, 0, stream>>>(seq, t + 1, cur);
    } else {
      const int writeOut = (t >= TINIT) ? 1 : 0;
      const int tGen = writeOut ? (t - TINIT) : 0;
      decoder_wmma<<<6, 256, 0, stream>>>(
          cur + KXP + 2 * HD, Wpd, biasd, out, cur + 0, writeOut, tGen);
    }
  }
}
